// GaussianMultinomialDiffusion_62886911148673
// MI455X (gfx1250) — matrix operand order repaired and checked by reading, not compiled, not run on hardware
//
#include <hip/hip_runtime.h>
#include <hip/hip_bf16.h>
#include <math.h>

typedef _Float16 h16;
typedef __attribute__((ext_vector_type(16))) _Float16 v16h;
typedef __attribute__((ext_vector_type(8)))  _Float16 v8h;
typedef __attribute__((ext_vector_type(8)))  float    v8f;
typedef unsigned int u32x4 __attribute__((ext_vector_type(4)));
typedef int          i32x4 __attribute__((ext_vector_type(4)));
typedef int          i32x8 __attribute__((ext_vector_type(8)));
typedef int          v4i   __attribute__((ext_vector_type(4)));

#define BATCH   16384
#define NNUM    16
#define NCAT    24
#define KCAT    16
#define CATDIM  384
#define DIN     400
#define DIN_PAD 416
#define DIMT    1024
#define NOUT_PAD 448
#define TSTEPS  1000

#ifndef __has_builtin
#define __has_builtin(x) 0
#endif

// ---- CDNA5 async global->LDS copy (ASYNCcnt path), with sync fallback ----
#if __has_builtin(__builtin_amdgcn_global_load_async_to_lds_b128)
#define HAVE_ASYNC 1
#else
#define HAVE_ASYNC 0
#endif

#if __has_builtin(__builtin_amdgcn_s_wait_asynccnt)
#define WAIT_ASYNC() __builtin_amdgcn_s_wait_asynccnt(0)
#else
#define WAIT_ASYNC() asm volatile("s_wait_asynccnt 0x0" ::: "memory")
#endif

// ---- CDNA5 Tensor Data Mover (TENSORcnt path) ----
#if __has_builtin(__builtin_amdgcn_tensor_load_to_lds)
#define HAVE_TDM 1
#else
#define HAVE_TDM 0
#endif

#if defined(__has_include)
#if __has_include(<hip/amd_detail/amd_gfx1250_TDM.h>)
#define TDM_6ARG 1
#else
#define TDM_6ARG 0
#endif
#else
#define TDM_6ARG 0
#endif

// LDS byte offset: generic LDS addresses carry the offset in the low 32 bits (ISA 10.2)
__device__ __forceinline__ unsigned int lds_off(const void* p) {
    return (unsigned int)(uintptr_t)p;
}

__device__ __forceinline__ void cp16_to_lds(void* ldst, const void* gsrc) {
#if HAVE_ASYNC
    // param0: AS1 (prints as __device__) pointer to int4; param1: AS3 pointer to int4
    __builtin_amdgcn_global_load_async_to_lds_b128(
        (__attribute__((address_space(1))) v4i*)(uintptr_t)gsrc,
        (__attribute__((address_space(3))) v4i*)(uintptr_t)lds_off(ldst),
        0, 0);
#else
    *(float4*)ldst = *(const float4*)gsrc;
#endif
}

#if HAVE_TDM
// 2D TDM load: tile_d0 units of 8B per row, tile_d1 rows, row stride = stride_u (8B units)
__device__ __forceinline__ void tdm_load_2d(unsigned int lds_addr, const void* gaddr,
                                            unsigned int tile_d0, unsigned int tile_d1,
                                            unsigned int stride_u, unsigned int rows) {
    unsigned long long ga = (unsigned long long)(uintptr_t)gaddr;
    u32x4 g0 = {0u, 0u, 0u, 0u};
    g0[0] = 1u;                                    // count=1, is_restore=0, gather=0
    g0[1] = lds_addr;                              // lds_addr
    g0[2] = (unsigned int)ga;                      // global_addr[31:0]
    g0[3] = (unsigned int)((ga >> 32) & 0x1FFFFFFu) | (2u << 30);  // addr[56:32] | type=2
    i32x8 g1 = {0, 0, 0, 0, 0, 0, 0, 0};
    g1[0] = (int)(3u << 16);                       // wg_mask=0, data_size=3 (8B)
    g1[1] = (int)((stride_u & 0xFFFFu) << 16);     // tensor_dim0[15:0]  (= row length)
    g1[2] = (int)((stride_u >> 16) & 0xFFFFu) | (int)((rows & 0xFFFFu) << 16); // td0 hi | td1 lo
    g1[3] = (int)((rows >> 16) & 0xFFFFu) | (int)((tile_d0 & 0xFFFFu) << 16);  // td1 hi | tile_dim0
    g1[4] = (int)(tile_d1 & 0xFFFFu);              // tile_dim1 (tile_dim2=0)
    g1[5] = (int)stride_u;                         // tensor_dim0_stride[31:0]
    i32x4 z4 = {0, 0, 0, 0};
#if TDM_6ARG
    i32x8 z8 = {0, 0, 0, 0, 0, 0, 0, 0};
    __builtin_amdgcn_tensor_load_to_lds(g0, g1, z4, z4, z8, 0);
#else
    __builtin_amdgcn_tensor_load_to_lds(g0, g1, z4, z4, 0);
#endif
}
#endif

__device__ __forceinline__ float logaddexpf(float a, float b) {
    float m = fmaxf(a, b);
    return m + logf(expf(a - m) + expf(b - m));
}

#define LOG_EPS (-69.07755278982137f)   // log(1e-30)
#define LOGK    (2.772588722239781f)    // log(16)

// ---------------- workspace layout (bytes) ----------------
constexpr size_t OFF_SCHED = 0;                                      // 6*1024 f32
constexpr size_t OFF_SAMP  = 24576;                                  // B*24 int
constexpr size_t OFF_WT1   = OFF_SAMP  + (size_t)BATCH * NCAT * 4;
constexpr size_t OFF_WT2   = OFF_WT1   + (size_t)1024 * 1024 * 2;
constexpr size_t OFF_WPROJ = OFF_WT2   + (size_t)1024 * 1024 * 2;    // [1024][416]
constexpr size_t OFF_WM1   = OFF_WPROJ + (size_t)1024 * DIN_PAD * 2;
constexpr size_t OFF_WM2   = OFF_WM1   + (size_t)1024 * 1024 * 2;    // [448][1024]
constexpr size_t OFF_XIN   = OFF_WM2   + (size_t)NOUT_PAD * 1024 * 2;
constexpr size_t OFF_A     = OFF_XIN   + (size_t)BATCH * DIN_PAD * 2;
constexpr size_t OFF_B     = OFF_A     + (size_t)BATCH * DIMT * 2;
constexpr size_t OFF_C     = OFF_B     + (size_t)BATCH * DIMT * 2;
constexpr size_t OFF_OUT   = OFF_C     + (size_t)BATCH * DIMT * 2;   // B*448 f32

// ---------------- schedule init (serial, tiny) ----------------
__global__ void sched_init(float* sch) {
    if (threadIdx.x != 0 || blockIdx.x != 0) return;
    const double PI = 3.14159265358979323846;
    double lca = 0.0;
    for (int i = 0; i < TSTEPS; i++) {
        double u1 = (((double)(i + 1) / TSTEPS) + 0.008) / 1.008 * PI * 0.5;
        double u0 = (((double)i / TSTEPS) + 0.008) / 1.008 * PI * 0.5;
        double ab1 = cos(u1); ab1 *= ab1;
        double ab0 = cos(u0); ab0 *= ab0;
        double beta = 1.0 - ab1 / ab0;
        if (beta > 0.999) beta = 0.999;
        double alpha = 1.0 - beta;
        double la = log(alpha);
        lca += la;
        double ac = exp(lca);
        sch[i]          = (float)la;                       // LOG_A
        sch[1024 + i]   = (float)log(1.0 - alpha + 1e-40); // LOG_1M_A
        sch[2048 + i]   = (float)lca;                      // LOG_CA
        sch[3072 + i]   = (float)log(1.0 - ac + 1e-40);    // LOG_1M_CA
        sch[4096 + i]   = (float)sqrt(ac);                 // SQRT_AC
        sch[5120 + i]   = (float)sqrt(1.0 - ac);           // SQRT_1M_AC
    }
}

// ---------------- weight convert: f32 [K][N] -> f16 transposed [Npad][Kpad] ----------------
__global__ void wt_convert(const float* __restrict__ W, h16* __restrict__ Wt,
                           int Kd, int N, int Kpad, int Npad) {
    int i = blockIdx.x * 256 + threadIdx.x;
    if (i >= Kpad * Npad) return;
    int k = i % Kpad;
    int n = i / Kpad;
    float v = (k < Kd && n < N) ? W[(size_t)k * N + n] : 0.f;
    Wt[(size_t)n * Kpad + k] = (h16)v;
}

// ---------------- per-sample prep: x_in, timestep embedding, gumbel sample ----------------
__global__ void prep_kernel(const float* __restrict__ x_num, const int* __restrict__ x_cat,
                            const int* __restrict__ t_arr, const float* __restrict__ noise,
                            const float* __restrict__ gum, const float* __restrict__ sch,
                            h16* __restrict__ xin, h16* __restrict__ te, int* __restrict__ samp) {
    int m = blockIdx.x;
    int tid = threadIdx.x;
    int t = t_arr[m];
    float lca_t = sch[2048 + t], l1mca_t = sch[3072 + t];
    float tf = (float)t;
    for (int i = 0; i < 4; i++) {
        int d = tid + i * 256;
        int j = d & 511;
        float f = expf(-9.210340371976184f * (float)j / 512.0f);
        float arg = tf * f;
        te[(size_t)m * DIMT + d] = (h16)((d < 512) ? cosf(arg) : sinf(arg));
    }
    if (tid < NNUM) {
        float v = sch[4096 + t] * x_num[m * NNUM + tid] + sch[5120 + t] * noise[m * NNUM + tid];
        xin[(size_t)m * DIN_PAD + tid] = (h16)v;
    } else if (tid >= 32 && tid < 32 + NCAT) {
        int c = tid - 32;
        int idx = x_cat[m * NCAT + c];
        float b = l1mca_t - LOGK;
        int best = 0;
        float bestv = -1e30f;
        for (int k = 0; k < KCAT; k++) {
            float u = gum[(size_t)m * CATDIM + c * KCAT + k];
            float g = -logf(-logf(u + 1e-30f) + 1e-30f);
            float a = ((k == idx) ? 0.f : LOG_EPS) + lca_t;
            float s = g + logaddexpf(a, b);
            if (s > bestv) { bestv = s; best = k; }
        }
        samp[m * NCAT + c] = best;
        for (int k = 0; k < KCAT; k++)
            xin[(size_t)m * DIN_PAD + NNUM + c * KCAT + k] = (h16)((k == best) ? 0.f : LOG_EPS);
    } else if (tid >= 64 && tid < 80) {
        xin[(size_t)m * DIN_PAD + DIN + (tid - 64)] = (h16)0.f;   // K padding
    }
}

// ---------------- WMMA GEMM: out = act(A @ W + bias [+ extra]) ----------------
// A: [M][lda] f16 row-major.  Wt: [N][ldk] f16 (W transposed).  128x64 tile per block (8 waves).
// Double-buffered: A tiles via GLOBAL_LOAD_ASYNC_TO_LDS_B128, B tiles via TDM tensor_load_to_lds.
template <int ACT, bool HAS_EXTRA, bool OUT_HALF>
__global__ __launch_bounds__(256) void gemm_wmma(
    const h16* __restrict__ A, int lda,
    const h16* __restrict__ Wt, int ldk,
    const float* __restrict__ bias, int nbias,
    const h16* __restrict__ extra, int lde,
    h16* __restrict__ outH, float* __restrict__ outF, int ldo,
    int Kd)
{
    __shared__ alignas(16) h16 lA[2][128 * 32];
    __shared__ alignas(16) h16 lB[2][64 * 32];
    const int tid   = threadIdx.x;
    const int lane  = tid & 31;
    const int w     = tid >> 5;
    const int rowBlk = blockIdx.y * 128;
    const int colBlk = blockIdx.x * 64;
    const int mloc  = lane & 15;
    const int khalf = lane >> 4;

    v8f acc[4];
    const v8f vz = {0.f, 0.f, 0.f, 0.f, 0.f, 0.f, 0.f, 0.f};
#pragma unroll
    for (int i = 0; i < 4; i++) acc[i] = vz;

    // A-tile chunk assignment: each thread moves two 16B chunks (128 rows x 4 chunks)
    const int ca0 = tid * 2;
    const int ra0 = ca0 >> 2, ka0 = (ca0 & 3) * 8;
    const int ra1 = (ca0 + 1) >> 2, ka1 = ((ca0 + 1) & 3) * 8;

    auto issueA = [&](int k0, int b) {
        cp16_to_lds(&lA[b][ra0 * 32 + ka0], A + (size_t)(rowBlk + ra0) * lda + k0 + ka0);
        cp16_to_lds(&lA[b][ra1 * 32 + ka1], A + (size_t)(rowBlk + ra1) * lda + k0 + ka1);
    };
    auto issueB = [&](int k0, int b) {
#if HAVE_TDM
        if (w == 0)   // one TDM DMA moves the whole 64x(32 half) = 4KB weight tile
            tdm_load_2d(lds_off(&lB[b][0]), Wt + (size_t)colBlk * ldk + k0,
                        /*tile_d0(8B units)=*/8, /*tile_d1=*/64,
                        /*stride(8B units)=*/(unsigned int)(ldk >> 2), /*rows=*/64u);
#else
        int row = tid >> 2, kc = (tid & 3) * 8;
        cp16_to_lds(&lB[b][row * 32 + kc], Wt + (size_t)(colBlk + row) * ldk + k0 + kc);
#endif
    };

    int buf = 0;
    issueA(0, 0);
    issueB(0, 0);

    for (int k0 = 0; k0 < Kd; k0 += 32) {
#if HAVE_TDM
        if (w == 0) __builtin_amdgcn_s_wait_tensorcnt(0);
#endif
        WAIT_ASYNC();
        __syncthreads();
        if (k0 + 32 < Kd) {
            issueA(k0 + 32, buf ^ 1);
            issueB(k0 + 32, buf ^ 1);
            __builtin_prefetch(A + (size_t)(rowBlk + (tid >> 1)) * lda + k0 + 64, 0, 1);
        }

        const h16* la = lA[buf];
        const h16* lb = lB[buf];
        // A fragment (ISA 7.12.2: lanes 0-15 kbase=0, lanes 16-31 kbase=8; chunk1 at K+16)
        union { v16h v; v8h h[2]; } af;
        const h16* ap = la + (w * 16 + mloc) * 32 + khalf * 8;
        af.h[0] = *(const v8h*)(ap);
        af.h[1] = *(const v8h*)(ap + 16);
#pragma unroll
        for (int tn = 0; tn < 4; tn++) {
            // B fragment: lanes 0-15 hold K=0..7,16..23, lanes 16-31 K=8..15,24..31, N=lane&15
            union { v16h v; v8h h[2]; } bf;
            const h16* bp = lb + (tn * 16 + mloc) * 32 + khalf * 8;
            bf.h[0] = *(const v8h*)(bp);
            bf.h[1] = *(const v8h*)(bp + 16);
            acc[tn] = __builtin_amdgcn_wmma_f32_16x16x32_f16(
                false, af.v, false, bf.v, (short)0, acc[tn], false, false);
        }
        buf ^= 1;
    }

    // epilogue: C layout — lane n = lane&15 (+tile), VGPR r -> M = r + (lane>=16)*8
#pragma unroll
    for (int tn = 0; tn < 4; tn++) {
        int n = colBlk + tn * 16 + mloc;
        float bi = (n < nbias) ? bias[n] : 0.f;
#pragma unroll
        for (int r = 0; r < 8; r++) {
            int mrow = rowBlk + w * 16 + khalf * 8 + r;
            float v = acc[tn][r] + bi;
            if (HAS_EXTRA) v += (float)extra[(size_t)mrow * lde + n];
            if (ACT == 1) v = fmaxf(v, 0.f);
            else if (ACT == 2) v = v / (1.f + __expf(-v));          // silu
            if (OUT_HALF) outH[(size_t)mrow * ldo + n] = (h16)v;
            else          outF[(size_t)mrow * ldo + n] = v;
        }
    }
}

// ---------------- loss ----------------
__global__ void zero_out(float* p) {
    if (threadIdx.x == 0 && blockIdx.x == 0) p[0] = 0.f;
}

__global__ void loss_kernel(const float* __restrict__ out, const float* __restrict__ noise,
                            const int* __restrict__ x_cat, const int* __restrict__ t_arr,
                            const int* __restrict__ samp, const float* __restrict__ sch,
                            float* __restrict__ d_out) {
    int m = blockIdx.x * 256 + threadIdx.x;
    float total = 0.f;
    if (m < BATCH) {
        int t = t_arr[m];
        const float* row = out + (size_t)m * NOUT_PAD;
        float g = 0.f;
        for (int j = 0; j < NNUM; j++) {
            float d = noise[m * NNUM + j] - row[j];
            g += d * d;
        }
        g *= (1.f / NNUM);
        int tm1 = (t > 0) ? t - 1 : 0;
        float la_t = sch[t],           l1ma_t  = sch[1024 + t];
        float lca_m = sch[2048 + tm1], l1mca_m = sch[3072 + tm1];
        float lcaT = sch[2048 + 999],  l1mcaT = sch[3072 + 999];
        float bqp = l1mca_m - LOGK;
        float bq1 = l1ma_t - LOGK;
        float bqT = l1mcaT - LOGK;
        float kl = 0.f, nll = 0.f, klp = 0.f;
        for (int c = 0; c < NCAT; c++) {
            const float* oc = row + NNUM + c * KCAT;
            int idx = x_cat[m * NCAT + c];
            int sp  = samp[m * NCAT + c];
            float mx = -1e30f;
            for (int k = 0; k < KCAT; k++) mx = fmaxf(mx, oc[k]);
            float s = 0.f;
            for (int k = 0; k < KCAT; k++) s += expf(oc[k] - mx);
            float lse = mx + logf(s);
            float unm[KCAT], unt[KCAT];
            for (int k = 0; k < KCAT; k++) {
                float lx0 = oc[k] - lse;
                float lev  = (t == 0) ? lx0 : logaddexpf(lx0 + lca_m, bqp);
                float lxt  = (k == sp) ? 0.f : LOG_EPS;
                float q1   = logaddexpf(lxt + la_t, bq1);
                unm[k] = lev + q1;
                float lx   = (k == idx) ? 0.f : LOG_EPS;
                float levt = (t == 0) ? lx : logaddexpf(lx + lca_m, bqp);
                unt[k] = levt + q1;
            }
            float m1 = -1e30f, m2 = -1e30f;
            for (int k = 0; k < KCAT; k++) { m1 = fmaxf(m1, unm[k]); m2 = fmaxf(m2, unt[k]); }
            float s1 = 0.f, s2 = 0.f;
            for (int k = 0; k < KCAT; k++) { s1 += expf(unm[k] - m1); s2 += expf(unt[k] - m2); }
            float lsem = m1 + logf(s1), lset = m2 + logf(s2);
            for (int k = 0; k < KCAT; k++) {
                float lmp = unm[k] - lsem;
                float ltp = unt[k] - lset;
                kl += expf(ltp) * (ltp - lmp);
                if (k == idx) nll -= lmp;
                float lqT = logaddexpf(((k == idx) ? 0.f : LOG_EPS) + lcaT, bqT);
                klp += expf(lqT) * (lqT + LOGK);
            }
        }
        float Lt = (t == 0) ? nll : kl;
        total = (Lt + klp) * (1.f / NCAT) + g;
    }
    __shared__ float red[256];
    red[threadIdx.x] = total;
    __syncthreads();
    for (int s = 128; s > 0; s >>= 1) {
        if (threadIdx.x < s) red[threadIdx.x] += red[threadIdx.x + s];
        __syncthreads();
    }
    if (threadIdx.x == 0) atomicAdd(d_out, red[0] * (1.f / BATCH));
}

// ---------------- launch ----------------
extern "C" void kernel_launch(void* const* d_in, const int* in_sizes, int n_in,
                              void* d_out, int out_size, void* d_ws, size_t ws_size,
                              hipStream_t stream) {
    const float* x_num   = (const float*)d_in[0];
    const int*   x_cat   = (const int*)d_in[1];
    const int*   t_arr   = (const int*)d_in[2];
    const float* noise   = (const float*)d_in[3];
    const float* gum     = (const float*)d_in[4];
    const float* te_w1   = (const float*)d_in[5];
    const float* te_b1   = (const float*)d_in[6];
    const float* te_w2   = (const float*)d_in[7];
    const float* te_b2   = (const float*)d_in[8];
    const float* proj_w  = (const float*)d_in[9];
    const float* proj_b  = (const float*)d_in[10];
    const float* mlp_w1  = (const float*)d_in[11];
    const float* mlp_b1  = (const float*)d_in[12];
    const float* mlp_w2  = (const float*)d_in[13];
    const float* mlp_b2  = (const float*)d_in[14];
    (void)in_sizes; (void)n_in; (void)ws_size;

    char* ws = (char*)d_ws;
    float* sch  = (float*)(ws + OFF_SCHED);
    int*   samp = (int*)  (ws + OFF_SAMP);
    h16*   wt1  = (h16*)  (ws + OFF_WT1);
    h16*   wt2  = (h16*)  (ws + OFF_WT2);
    h16*   wpj  = (h16*)  (ws + OFF_WPROJ);
    h16*   wm1  = (h16*)  (ws + OFF_WM1);
    h16*   wm2  = (h16*)  (ws + OFF_WM2);
    h16*   xin  = (h16*)  (ws + OFF_XIN);
    h16*   bufA = (h16*)  (ws + OFF_A);
    h16*   bufB = (h16*)  (ws + OFF_B);
    h16*   bufC = (h16*)  (ws + OFF_C);
    float* mout = (float*)(ws + OFF_OUT);
    float* loss = (float*)d_out;
    (void)out_size;

    zero_out<<<1, 64, 0, stream>>>(loss);
    sched_init<<<1, 32, 0, stream>>>(sch);

    wt_convert<<<(1024 * 1024 + 255) / 256, 256, 0, stream>>>(te_w1, wt1, 1024, 1024, 1024, 1024);
    wt_convert<<<(1024 * 1024 + 255) / 256, 256, 0, stream>>>(te_w2, wt2, 1024, 1024, 1024, 1024);
    wt_convert<<<(1024 * DIN_PAD + 255) / 256, 256, 0, stream>>>(proj_w, wpj, DIN, 1024, DIN_PAD, 1024);
    wt_convert<<<(1024 * 1024 + 255) / 256, 256, 0, stream>>>(mlp_w1, wm1, 1024, 1024, 1024, 1024);
    wt_convert<<<(NOUT_PAD * 1024 + 255) / 256, 256, 0, stream>>>(mlp_w2, wm2, 1024, DIN, 1024, NOUT_PAD);

    prep_kernel<<<BATCH, 256, 0, stream>>>(x_num, x_cat, t_arr, noise, gum, sch, xin, bufA, samp);

    dim3 blk(256);
    dim3 gFull(DIMT / 64, BATCH / 128);     // N=1024
    dim3 gOut(NOUT_PAD / 64, BATCH / 128);  // N=448

    // emb_h = silu(te @ te_w1 + te_b1)
    gemm_wmma<2, false, true><<<gFull, blk, 0, stream>>>(
        bufA, DIMT, wt1, 1024, te_b1, 1024, (const h16*)nullptr, 0, bufB, (float*)nullptr, DIMT, 1024);
    // emb = emb_h @ te_w2 + te_b2
    gemm_wmma<0, false, true><<<gFull, blk, 0, stream>>>(
        bufB, DIMT, wt2, 1024, te_b2, 1024, (const h16*)nullptr, 0, bufC, (float*)nullptr, DIMT, 1024);
    // h = x_in @ proj_w + proj_b + emb
    gemm_wmma<0, true, true><<<gFull, blk, 0, stream>>>(
        xin, DIN_PAD, wpj, DIN_PAD, proj_b, 1024, bufC, DIMT, bufA, (float*)nullptr, DIMT, DIN_PAD);
    // h1 = relu(h @ mlp_w1 + mlp_b1)
    gemm_wmma<1, false, true><<<gFull, blk, 0, stream>>>(
        bufA, DIMT, wm1, 1024, mlp_b1, 1024, (const h16*)nullptr, 0, bufB, (float*)nullptr, DIMT, 1024);
    // model_out = h1 @ mlp_w2 + mlp_b2   (f32, N padded to 448, bias valid for n<400)
    gemm_wmma<0, false, false><<<gOut, blk, 0, stream>>>(
        bufB, DIMT, wm2, 1024, mlp_b2, DIN, (const h16*)nullptr, 0, (h16*)nullptr, mout, NOUT_PAD, 1024);

    loss_kernel<<<BATCH / 256, 256, 0, stream>>>(mout, noise, x_cat, t_arr, samp, sch, loss);
}
